// kmeans_24532853195390
// MI455X (gfx1250) — matrix operand order repaired and checked by reading, not compiled, not run on hardware
//
#include <hip/hip_runtime.h>

// K-means assignment:  out[n] = argmin_c ( p2[c] - 2 * X[n,:] . Phi[c,:] )
// (the ||x||^2 term is row-constant and dropped from the argmin)
//
// Strategy: split-bf16 (hi+lo) GEMM on v_wmma_f32_16x16x32_bf16, Phi staged
// in LDS, per-wave 16-row tiles, cross-lane argmin reduction.

#define NPTS 1000000
#define DIM  50
#define NC   256
#define KP   64               // K padded to 2 x 32 for the bf16 WMMA
#define ROWS_PER_BLOCK 128    // 8 waves * 16 rows
#define THREADS 256

typedef __attribute__((ext_vector_type(16))) __bf16 v16bf;
typedef __attribute__((ext_vector_type(8)))  float  v8f;

union V16 {
  v16bf v;
  uint4 q[2];
  unsigned short u[16];
};

static __device__ __forceinline__ unsigned short f2bf(float f) {
  // round-to-nearest-even fp32 -> bf16 (inputs are finite normals)
  unsigned int x = __float_as_uint(f);
  unsigned int r = x + 0x7FFFu + ((x >> 16) & 1u);
  return (unsigned short)(r >> 16);
}
static __device__ __forceinline__ float bf2f(unsigned short u) {
  return __uint_as_float(((unsigned int)u) << 16);
}

// --- prep: split Phi into bf16 hi/lo (padded to KP) and compute ||phi||^2 ---
__global__ void prep_phi_kernel(const float* __restrict__ Phi,
                                unsigned short* __restrict__ phiHi,
                                unsigned short* __restrict__ phiLo,
                                float* __restrict__ p2) {
  int c = threadIdx.x;
  if (c >= NC) return;
  const float* row = Phi + c * DIM;
  float s = 0.f;
  for (int k = 0; k < KP; ++k) {
    float v = (k < DIM) ? row[k] : 0.f;
    s += v * v;
    unsigned short hi = f2bf(v);
    unsigned short lo = f2bf(v - bf2f(hi));
    phiHi[c * KP + k] = hi;
    phiLo[c * KP + k] = lo;
  }
  p2[c] = s;
}

// --- main: one wave = 16 rows of X vs all 256 centroids ---
__global__ void __launch_bounds__(THREADS)
assign_kernel(const float* __restrict__ X,
              const unsigned short* __restrict__ phiHi,
              const unsigned short* __restrict__ phiLo,
              const float* __restrict__ p2,
              int* __restrict__ out) {
  extern __shared__ char smem[];
  unsigned short* sHi = (unsigned short*)smem;                    // NC*KP bf16
  unsigned short* sLo = (unsigned short*)(smem + NC * KP * 2);    // NC*KP bf16
  float*          sP2 = (float*)(smem + NC * KP * 4);             // NC floats

  // cooperative LDS fill (Phi stays L2-resident; one 65KB pull per block)
  {
    const uint4* gHi = (const uint4*)phiHi;
    const uint4* gLo = (const uint4*)phiLo;
    uint4* dHi = (uint4*)sHi;
    uint4* dLo = (uint4*)sLo;
    const int nvec = (NC * KP * 2) / 16;   // 2048 uint4 per array
    for (int i = threadIdx.x; i < nvec; i += THREADS) {
      dHi[i] = gHi[i];
      dLo[i] = gLo[i];
    }
    if (threadIdx.x < NC) sP2[threadIdx.x] = p2[threadIdx.x];
  }
  __syncthreads();

  const int tid  = threadIdx.x;
  const int lane = tid & 31;       // wave32
  const int wid  = tid >> 5;
  const int l16  = lane & 15;
  const int hiH  = lane >> 4;      // 0 = lanes 0-15, 1 = lanes 16-31

  const long rowBase = (long)blockIdx.x * ROWS_PER_BLOCK + wid * 16;
  int arow = (int)rowBase + l16;
  if (arow >= NPTS) arow = NPTS - 1;               // clamp; stores are guarded
  const float* xr = X + (long)arow * DIM;

  // A operand, contiguous k order used here (not the ISA table order; B matches):
  //  lane L (<16): row L, K 0..15 of the k-step (contiguous)
  //  lane L+16   : row L, K 16..31 of the k-step
  V16 aHi[2], aLo[2];
#pragma unroll
  for (int ks = 0; ks < 2; ++ks) {
    const int k0 = ks * 32 + hiH * 16;
    const int k1 = k0 + 8;
#pragma unroll
    for (int i = 0; i < 8; ++i) {
      float f0 = (k0 + i < DIM) ? xr[k0 + i] : 0.f;
      float f1 = (k1 + i < DIM) ? xr[k1 + i] : 0.f;
      unsigned short h0 = f2bf(f0);
      unsigned short h1 = f2bf(f1);
      aHi[ks].u[i]     = h0;
      aHi[ks].u[8 + i] = h1;
      aLo[ks].u[i]     = f2bf(f0 - bf2f(h0));
      aLo[ks].u[8 + i] = f2bf(f1 - bf2f(h1));
    }
  }

  float best[8];
  int   bidx[8];
#pragma unroll
  for (int j = 0; j < 8; ++j) { best[j] = 3.4e38f; bidx[j] = 0; }

  for (int t = 0; t < NC / 16; ++t) {
    const int c = t * 16 + l16;      // this lane's centroid column
    v8f acc = {};
#pragma unroll
    for (int ks = 0; ks < 2; ++ks) {
      // B operand, same contiguous k order as A: lanes 0-15 hold K 0..15 (of this k-step),
      // lanes 16-31 hold K 16..31; column n = centroid c0+n.
      const int kB = ks * 32 + hiH * 16;
      V16 bh, bl;
      const uint4* ph = (const uint4*)(sHi + c * KP + kB);
      const uint4* pl = (const uint4*)(sLo + c * KP + kB);
      bh.q[0] = ph[0]; bh.q[1] = ph[1];
      bl.q[0] = pl[0]; bl.q[1] = pl[1];
      // split product: hi*hi + hi*lo + lo*hi  (lo*lo ~ 2^-16, dropped)
      acc = __builtin_amdgcn_wmma_f32_16x16x32_bf16(
          false, aHi[ks].v, false, bh.v, (short)0, acc, false, false);
      acc = __builtin_amdgcn_wmma_f32_16x16x32_bf16(
          false, aHi[ks].v, false, bl.v, (short)0, acc, false, false);
      acc = __builtin_amdgcn_wmma_f32_16x16x32_bf16(
          false, aLo[ks].v, false, bh.v, (short)0, acc, false, false);
    }
    const float pc = sP2[c];
#pragma unroll
    for (int j = 0; j < 8; ++j) {
      float val = pc - 2.f * acc[j];   // d^2 - ||x||^2 (row-constant dropped)
      if (val < best[j]) { best[j] = val; bidx[j] = c; }   // first-min kept
    }
  }

  // reduce argmin across the 16 lanes sharing each row (masks stay in-half)
#pragma unroll
  for (int j = 0; j < 8; ++j) {
#pragma unroll
    for (int m = 1; m < 16; m <<= 1) {
      float of = __shfl_xor(best[j], m, 32);
      int   oi = __shfl_xor(bidx[j], m, 32);
      if (of < best[j] || (of == best[j] && oi < bidx[j])) {
        best[j] = of; bidx[j] = oi;
      }
    }
  }

  // lane l16==0 of each half holds rows M = hiH*8 + j
  if (l16 == 0) {
#pragma unroll
    for (int j = 0; j < 8; ++j) {
      long r = rowBase + hiH * 8 + j;
      if (r < NPTS) out[r] = bidx[j];
    }
  }
}

extern "C" void kernel_launch(void* const* d_in, const int* in_sizes, int n_in,
                              void* d_out, int out_size, void* d_ws, size_t ws_size,
                              hipStream_t stream) {
  const float* X   = (const float*)d_in[0];   // [N, D] fp32
  const float* Phi = (const float*)d_in[1];   // [C, D] fp32
  int* out = (int*)d_out;                     // [N] int32

  // workspace layout: phiHi[NC*KP] u16 | phiLo[NC*KP] u16 | p2[NC] f32
  unsigned short* phiHi = (unsigned short*)d_ws;
  unsigned short* phiLo = phiHi + NC * KP;
  float* p2 = (float*)(phiLo + NC * KP);

  prep_phi_kernel<<<1, 256, 0, stream>>>(Phi, phiHi, phiLo, p2);

  const int grid = (NPTS + ROWS_PER_BLOCK - 1) / ROWS_PER_BLOCK;
  const size_t shmem = (size_t)NC * KP * 4 + (size_t)NC * 4;  // 66560 B
  assign_kernel<<<grid, THREADS, shmem, stream>>>(X, phiHi, phiLo, p2, out);
}
